// MultiHeadAttention_16793322127585
// MI455X (gfx1250) — matrix operand order repaired and checked by reading, not compiled, not run on hardware
//
#include <hip/hip_runtime.h>

// ---------------------------------------------------------------------------
// MHA forward for MI455X (gfx1250): bf16 WMMA, fp32 accumulate.
// B=8, N=1024, D=768, H=12, hd=64.
// Round 7: explicit amdgpu_waves_per_eu(2) so the register allocator budgets
// >=256 VGPRs (kills the 2x uint4 scratch spill seen in round 6).
// ---------------------------------------------------------------------------

typedef __attribute__((ext_vector_type(16))) __bf16 v16bf;
typedef __attribute__((ext_vector_type(8)))  float  v8f;
typedef __attribute__((ext_vector_type(2)))  float  f32x2;
typedef __attribute__((ext_vector_type(2)))  __bf16 bf16x2;

union BFrag {
    v16bf v;
    unsigned int u[8];
    uint4 q[2];
};

__device__ __forceinline__ v8f vzero8() {
    v8f z;
#pragma unroll
    for (int i = 0; i < 8; ++i) z[i] = 0.0f;
    return z;
}

// packed fp32x4 -> bf16x4 (2x v_cvt_pk_bf16_f32)
__device__ __forceinline__ unsigned long long pack4bf(float4 f) {
    union { bf16x2 h2[2]; unsigned long long u; } t;
    t.h2[0] = __builtin_convertvector((f32x2){f.x, f.y}, bf16x2);
    t.h2[1] = __builtin_convertvector((f32x2){f.z, f.w}, bf16x2);
    return t.u;
}

// ===========================================================================
// Kernel 0: transpose + convert weights: w[K][N] fp32 -> wt[N][K] bf16.
// ===========================================================================
__global__ __launch_bounds__(256) void wtrans_kernel(
    const float* __restrict__ w, __bf16* __restrict__ wt, int K, int N)
{
    __shared__ __bf16 tile[32 * 33];
    const int tid = threadIdx.x;
    const int n0 = blockIdx.x * 32;
    const int k0 = blockIdx.y * 32;
#pragma unroll
    for (int i = 0; i < 4; ++i) {
        int f = tid + 256 * i;
        int kk = f >> 5, nn = f & 31;
        tile[kk * 33 + nn] = (__bf16)w[(size_t)(k0 + kk) * N + n0 + nn];
    }
    __syncthreads();
#pragma unroll
    for (int i = 0; i < 4; ++i) {
        int f = tid + 256 * i;
        int nn = f >> 5, kk = f & 31;
        wt[(size_t)(n0 + nn) * K + k0 + kk] = tile[kk * 33 + nn];
    }
}

// ===========================================================================
// Kernel 1: qkv = x @ w_qkv + b_qkv -> Q(*1/8),K in [b,h,n,d]; V in [b,h,d,n].
// Block 128 (4 waves), tile 128x64, 32 rows/wave, K-step 32, double-buffered,
// B-fragment shared by 2 WMMAs.
// ===========================================================================
__global__ __launch_bounds__(128)
__attribute__((amdgpu_waves_per_eu(2)))
void qkv_gemm_kernel(
    const float* __restrict__ x,       // [8192, 768]
    const __bf16* __restrict__ wt,     // [2304, 768]  (K-major)
    const float* __restrict__ bias,    // [2304]
    __bf16* __restrict__ qo,           // [96, 1024, 64] (pre-scaled by 1/8)
    __bf16* __restrict__ ko,           // [96, 1024, 64]
    __bf16* __restrict__ vto)          // [96, 64, 1024] (transposed)
{
    __shared__ __bf16 lA[2][128 * 40];   // A tile [128][32], 80 B rows
    __shared__ __bf16 lBt[2][64 * 40];   // B tile K-major [64 n][32 k]

    const int tid  = threadIdx.x;
    const int wave = tid >> 5;
    const int lane = tid & 31;
    const int l16  = lane & 15;
    const int half = lane >> 4;
    const int row0 = blockIdx.x * 128;
    const int col0 = blockIdx.y * 64;

    // staging bases: chunk i is a constant offset from these
    const float*  xp  = x  + (size_t)(row0 + (tid >> 3)) * 768 + 4 * (tid & 7);
    __bf16*       la0 = &lA[0][(tid >> 3) * 40 + 4 * (tid & 7)];
    const __bf16* wp  = wt + (size_t)(col0 + (tid >> 2)) * 768 + 8 * (tid & 3);
    __bf16*       lb0 = &lBt[0][(tid >> 2) * 40 + 8 * (tid & 3)];

    unsigned long long pa64[8];
    uint4 qb[2];
#pragma unroll
    for (int i = 0; i < 8; ++i)
        pa64[i] = pack4bf(*(const float4*)(xp + i * (16 * 768)));
#pragma unroll
    for (int i = 0; i < 2; ++i)
        qb[i] = *(const uint4*)(wp + i * (32 * 768));

    v8f acc[2][4];
#pragma unroll
    for (int mg = 0; mg < 2; ++mg)
#pragma unroll
        for (int t = 0; t < 4; ++t) acc[mg][t] = vzero8();

    int p = 0;
    for (int k0 = 0; k0 < 768; k0 += 32) {
        __bf16* la = la0 + p * (128 * 40);
        __bf16* lb = lb0 + p * (64 * 40);
#pragma unroll
        for (int i = 0; i < 8; ++i)
            *(unsigned long long*)(la + i * (16 * 40)) = pa64[i];
#pragma unroll
        for (int i = 0; i < 2; ++i)
            *(uint4*)(lb + i * (32 * 40)) = qb[i];
        __syncthreads();

        if (k0 + 32 < 768) {
#pragma unroll
            for (int i = 0; i < 8; ++i)
                pa64[i] = pack4bf(*(const float4*)(xp + (k0 + 32) + i * (16 * 768)));
#pragma unroll
            for (int i = 0; i < 2; ++i)
                qb[i] = *(const uint4*)(wp + (k0 + 32) + i * (32 * 768));
        }

        BFrag af[2];
#pragma unroll
        for (int mg = 0; mg < 2; ++mg) {
            const __bf16* ap = &lA[p][(wave * 32 + mg * 16 + l16) * 40 + half * 8];
            af[mg].q[0] = *(const uint4*)(ap);
            af[mg].q[1] = *(const uint4*)(ap + 16);
        }
#pragma unroll
        for (int t = 0; t < 4; ++t) {
            BFrag b;   // one B fragment feeds both M-subtiles
            const __bf16* bp = &lBt[p][(t * 16 + l16) * 40 + half * 8];
            b.q[0] = *(const uint4*)(bp);
            b.q[1] = *(const uint4*)(bp + 16);
            acc[0][t] = __builtin_amdgcn_wmma_f32_16x16x32_bf16(
                false, af[0].v, false, b.v, (short)0, acc[0][t], false, false);
            acc[1][t] = __builtin_amdgcn_wmma_f32_16x16x32_bf16(
                false, af[1].v, false, b.v, (short)0, acc[1][t], false, false);
        }
        p ^= 1;
    }

    // epilogue: bias; Q (scaled 1/8), K -> [b,h,n,d]; V -> [b,h,d,n]
#pragma unroll
    for (int t = 0; t < 4; ++t) {
        int cg = col0 + t * 16 + l16;
        float bv = bias[cg];
        int sec = cg / 768;            // uniform per (block, t)
        int j = cg % 768;
        int h = j >> 6, d = j & 63;
#pragma unroll
        for (int mg = 0; mg < 2; ++mg)
#pragma unroll
            for (int r = 0; r < 8; ++r) {
                int rowg = row0 + wave * 32 + mg * 16 + r + 8 * half;
                int bb = rowg >> 10, n = rowg & 1023;
                float o = acc[mg][t][r] + bv;
                if (sec == 0)
                    qo[(((size_t)bb * 12 + h) * 1024 + n) * 64 + d] = (__bf16)(o * 0.125f);
                else if (sec == 1)
                    ko[(((size_t)bb * 12 + h) * 1024 + n) * 64 + d] = (__bf16)o;
                else
                    vto[(((size_t)bb * 12 + h) * 64 + d) * 1024 + n] = (__bf16)o;
            }
    }
}

// ===========================================================================
// Kernel 2: flash attention per (b*h, 64-query tile). Block: 128 threads.
// Register-space online softmax (shuffle butterflies), 2 barriers/iteration.
// ===========================================================================
__global__ __launch_bounds__(128)
__attribute__((amdgpu_waves_per_eu(2)))
void attn_kernel(
    const __bf16* __restrict__ q,     // [96, 1024, 64] (pre-scaled)
    const __bf16* __restrict__ k,     // [96, 1024, 64]
    const __bf16* __restrict__ vt,    // [96, 64, 1024]
    __bf16* __restrict__ o)           // [8, 1024, 768]
{
    __shared__ __bf16 lK[64 * 72];    // K tile [key][hd], 144 B rows
    __shared__ __bf16 lVt[64 * 72];   // V tile [hd][key]
    __shared__ __bf16 lP[64 * 72];    // probs bf16 [q][key]

    const int tid  = threadIdx.x;
    const int wave = tid >> 5;
    const int lane = tid & 31;
    const int l16  = lane & 15;
    const int half = lane >> 4;
    const int qt = blockIdx.x;        // 0..15
    const int bh = blockIdx.y;        // 0..95

    BFrag aqf[2];
    {
        const __bf16* qp =
            q + ((size_t)bh * 1024 + qt * 64 + wave * 16 + l16) * 64 + half * 8;
#pragma unroll
        for (int kc = 0; kc < 2; ++kc) {
            aqf[kc].q[0] = *(const uint4*)(qp + kc * 32);
            aqf[kc].q[1] = *(const uint4*)(qp + kc * 32 + 16);
        }
    }

    v8f accO[4];
#pragma unroll
    for (int t = 0; t < 4; ++t) accO[t] = vzero8();

    float mrun[8], lrun[8];
#pragma unroll
    for (int r = 0; r < 8; ++r) { mrun[r] = -3.0e38f; lrun[r] = 0.0f; }

    // staging bases (chunk i = constant offset)
    const int sr = tid >> 3;          // 0..15
    const int sc = 8 * (tid & 7);     // 0..56
    const __bf16* kp0  = k  + (size_t)bh * 1024 * 64 + (size_t)sr * 64 + sc;
    const __bf16* vtp0 = vt + (size_t)bh * 64 * 1024 + (size_t)sr * 1024 + sc;
    __bf16* lk0  = &lK[sr * 72 + sc];
    __bf16* lvt0 = &lVt[sr * 72 + sc];

    for (int jt = 0; jt < 16; ++jt) {
        __syncthreads();
        const __bf16* kp  = kp0  + jt * (64 * 64);
        const __bf16* vtp = vtp0 + jt * 64;
#pragma unroll
        for (int i = 0; i < 4; ++i) {
            *(uint4*)(lk0  + i * (16 * 72)) = *(const uint4*)(kp  + i * (16 * 64));
            *(uint4*)(lvt0 + i * (16 * 72)) = *(const uint4*)(vtp + i * (16 * 1024));
        }
        __syncthreads();

        // S = Q K^T (Q pre-scaled), 4 tiles in registers
        v8f st[4];
#pragma unroll
        for (int t = 0; t < 4; ++t) {
            st[t] = vzero8();
#pragma unroll
            for (int kc = 0; kc < 2; ++kc) {
                BFrag b;
                const __bf16* bp = &lK[(t * 16 + l16) * 72 + kc * 32 + half * 8];
                b.q[0] = *(const uint4*)(bp);
                b.q[1] = *(const uint4*)(bp + 16);
                st[t] = __builtin_amdgcn_wmma_f32_16x16x32_bf16(
                    false, aqf[kc].v, false, b.v, (short)0, st[t], false, false);
            }
        }

        // online softmax in registers
        float corr[8], mnew[8];
#pragma unroll
        for (int r = 0; r < 8; ++r) {
            float mx = fmaxf(fmaxf(st[0][r], st[1][r]), fmaxf(st[2][r], st[3][r]));
#pragma unroll
            for (int m = 1; m < 16; m <<= 1)
                mx = fmaxf(mx, __shfl_xor(mx, m, 32));
            mnew[r] = fmaxf(mrun[r], mx);
            corr[r] = __expf(mrun[r] - mnew[r]);
            mrun[r] = mnew[r];
        }
#pragma unroll
        for (int r = 0; r < 8; ++r) {
            int prow = (wave * 16 + r + 8 * half) * 72 + l16;
            float sum = 0.0f;
#pragma unroll
            for (int t = 0; t < 4; ++t) {
                float pv = __expf(st[t][r] - mnew[r]);
                sum += pv;
                lP[prow + t * 16] = (__bf16)pv;
            }
#pragma unroll
            for (int m = 1; m < 16; m <<= 1)
                sum += __shfl_xor(sum, m, 32);
            lrun[r] = lrun[r] * corr[r] + sum;
        }

        // O = O*corr + P @ V (wave reads only its own P rows; same-wave
        // DS ordering makes this safe without a barrier)
        BFrag ap2[2];
        const __bf16* pp = &lP[(wave * 16 + l16) * 72 + half * 8];
#pragma unroll
        for (int kc = 0; kc < 2; ++kc) {
            ap2[kc].q[0] = *(const uint4*)(pp + kc * 32);
            ap2[kc].q[1] = *(const uint4*)(pp + kc * 32 + 16);
        }
#pragma unroll
        for (int t = 0; t < 4; ++t) {
#pragma unroll
            for (int r = 0; r < 8; ++r) accO[t][r] *= corr[r];
#pragma unroll
            for (int kc = 0; kc < 2; ++kc) {
                BFrag b;
                const __bf16* bp = &lVt[(t * 16 + l16) * 72 + kc * 32 + half * 8];
                b.q[0] = *(const uint4*)(bp);
                b.q[1] = *(const uint4*)(bp + 16);
                accO[t] = __builtin_amdgcn_wmma_f32_16x16x32_bf16(
                    false, ap2[kc].v, false, b.v, (short)0, accO[t], false, false);
            }
        }
    }

    float inv[8];
#pragma unroll
    for (int r = 0; r < 8; ++r) inv[r] = 1.0f / lrun[r];
    const int bb = bh / 12, h = bh % 12;
#pragma unroll
    for (int t = 0; t < 4; ++t)
#pragma unroll
        for (int r = 0; r < 8; ++r) {
            int rowg = qt * 64 + wave * 16 + r + 8 * half;
            size_t idx = ((size_t)bb * 1024 + rowg) * 768 + h * 64 + t * 16 + l16;
            o[idx] = (__bf16)(accO[t][r] * inv[r]);
        }
}

// ===========================================================================
// Kernel 3: out = attn @ w_proj + b_proj (fp32 out). Block 128 (4 waves),
// tile 128x64, 32 rows/wave, double-buffered, B-fragment shared by 2 WMMAs.
// ===========================================================================
__global__ __launch_bounds__(128)
__attribute__((amdgpu_waves_per_eu(2)))
void proj_gemm_kernel(
    const __bf16* __restrict__ a,      // [8192, 768] bf16
    const __bf16* __restrict__ wt,     // [768, 768]  (K-major)
    const float* __restrict__ bias,    // [768]
    float* __restrict__ out)           // [8192, 768]
{
    __shared__ __bf16 lA[2][128 * 40];
    __shared__ __bf16 lBt[2][64 * 40];

    const int tid  = threadIdx.x;
    const int wave = tid >> 5;
    const int lane = tid & 31;
    const int l16  = lane & 15;
    const int half = lane >> 4;
    const int row0 = blockIdx.x * 128;
    const int col0 = blockIdx.y * 64;

    const __bf16* ap0 = a  + (size_t)(row0 + (tid >> 2)) * 768 + 8 * (tid & 3);
    __bf16*       la0 = &lA[0][(tid >> 2) * 40 + 8 * (tid & 3)];
    const __bf16* wp  = wt + (size_t)(col0 + (tid >> 2)) * 768 + 8 * (tid & 3);
    __bf16*       lb0 = &lBt[0][(tid >> 2) * 40 + 8 * (tid & 3)];

    uint4 qa[4], qb[2];
#pragma unroll
    for (int i = 0; i < 4; ++i)
        qa[i] = *(const uint4*)(ap0 + i * (32 * 768));
#pragma unroll
    for (int i = 0; i < 2; ++i)
        qb[i] = *(const uint4*)(wp + i * (32 * 768));

    v8f acc[2][4];
#pragma unroll
    for (int mg = 0; mg < 2; ++mg)
#pragma unroll
        for (int t = 0; t < 4; ++t) acc[mg][t] = vzero8();

    int p = 0;
    for (int k0 = 0; k0 < 768; k0 += 32) {
        __bf16* la = la0 + p * (128 * 40);
        __bf16* lb = lb0 + p * (64 * 40);
#pragma unroll
        for (int i = 0; i < 4; ++i)
            *(uint4*)(la + i * (32 * 40)) = qa[i];
#pragma unroll
        for (int i = 0; i < 2; ++i)
            *(uint4*)(lb + i * (32 * 40)) = qb[i];
        __syncthreads();

        if (k0 + 32 < 768) {
#pragma unroll
            for (int i = 0; i < 4; ++i)
                qa[i] = *(const uint4*)(ap0 + (k0 + 32) + i * (32 * 768));
#pragma unroll
            for (int i = 0; i < 2; ++i)
                qb[i] = *(const uint4*)(wp + (k0 + 32) + i * (32 * 768));
        }

        BFrag af[2];
#pragma unroll
        for (int mg = 0; mg < 2; ++mg) {
            const __bf16* app = &lA[p][(wave * 32 + mg * 16 + l16) * 40 + half * 8];
            af[mg].q[0] = *(const uint4*)(app);
            af[mg].q[1] = *(const uint4*)(app + 16);
        }
#pragma unroll
        for (int t = 0; t < 4; ++t) {
            BFrag b;
            const __bf16* bp = &lBt[p][(t * 16 + l16) * 40 + half * 8];
            b.q[0] = *(const uint4*)(bp);
            b.q[1] = *(const uint4*)(bp + 16);
            acc[0][t] = __builtin_amdgcn_wmma_f32_16x16x32_bf16(
                false, af[0].v, false, b.v, (short)0, acc[0][t], false, false);
            acc[1][t] = __builtin_amdgcn_wmma_f32_16x16x32_bf16(
                false, af[1].v, false, b.v, (short)0, acc[1][t], false, false);
        }
        p ^= 1;
    }

#pragma unroll
    for (int t = 0; t < 4; ++t) {
        int cg = col0 + t * 16 + l16;
        float bv = bias[cg];
#pragma unroll
        for (int mg = 0; mg < 2; ++mg)
#pragma unroll
            for (int r = 0; r < 8; ++r) {
                int rowg = row0 + wave * 32 + mg * 16 + r + 8 * half;
                out[(size_t)rowg * 768 + cg] = acc[mg][t][r] + bv;
            }
    }
}

// ===========================================================================
// Launch: wtrans x2 -> qkv gemm -> flash attn -> out proj
// ===========================================================================
extern "C" void kernel_launch(void* const* d_in, const int* in_sizes, int n_in,
                              void* d_out, int out_size, void* d_ws, size_t ws_size,
                              hipStream_t stream) {
    const float* x     = (const float*)d_in[0];
    const float* wqkv  = (const float*)d_in[1];
    const float* bqkv  = (const float*)d_in[2];
    const float* wproj = (const float*)d_in[3];
    const float* bproj = (const float*)d_in[4];
    float* out = (float*)d_out;

    const size_t elems = (size_t)8 * 12 * 1024 * 64;  // 6291456 per tensor
    __bf16* qws    = (__bf16*)d_ws;
    __bf16* kws    = qws + elems;
    __bf16* vtws   = kws + elems;
    __bf16* aws    = vtws + elems;
    __bf16* wqkvT  = aws + elems;                     // [2304][768]
    __bf16* wprojT = wqkvT + (size_t)2304 * 768;      // [768][768]

    wtrans_kernel<<<dim3(72, 24), 256, 0, stream>>>(wqkv, wqkvT, 768, 2304);
    wtrans_kernel<<<dim3(24, 24), 256, 0, stream>>>(wproj, wprojT, 768, 768);
    qkv_gemm_kernel<<<dim3(64, 36), 128, 0, stream>>>(x, wqkvT, bqkv, qws, kws, vtws);
    attn_kernel<<<dim3(16, 96), 128, 0, stream>>>(qws, kws, vtws, aws);
    proj_gemm_kernel<<<dim3(64, 12), 128, 0, stream>>>(aws, wprojT, bproj, out);
}
